// OutputBlock_26018911879799
// MI455X (gfx1250) — matrix operand order repaired and checked by reading, not compiled, not run on hardware
//
#include <hip/hip_runtime.h>
#include <hip/hip_bf16.h>

// ---------------- types ----------------
typedef __attribute__((ext_vector_type(16))) __bf16 v16bf;
typedef __attribute__((ext_vector_type(8)))  __bf16 v8bf;
typedef __attribute__((ext_vector_type(8)))  float  v8f;

#define B_   512
#define L_   5
#define M_   2560           // B*L
#define DZ_  32
#define E_   896            // sum DIM_LAT
#define D_   20640          // sum DIM_OUT
#define BN_EPS_ 0.001f
#define ALPHA_  0.3f

__device__ __forceinline__ __bf16 f2bf(float f) {
    unsigned int u = __builtin_bit_cast(unsigned int, f);
    unsigned int r = (u + 0x7FFFu + ((u >> 16) & 1u)) >> 16;
    unsigned short s = (unsigned short)r;
    return __builtin_bit_cast(__bf16, s);
}

__device__ __forceinline__ float logsig(float t) {
    // log(sigmoid(t)) = min(t,0) - log1p(exp(-|t|))
    return fminf(t, 0.0f) - log1pf(__expf(-fabsf(t)));
}

// ---------------- small kernels ----------------
__global__ void zero_kernel(float* out, int n) {
    int i = blockIdx.x * blockDim.x + threadIdx.x;
    if (i < n) out[i] = 0.0f;
}

// h = leakyrelu(z @ w1 + b1), written to hpre[:, offe:offe+dl] (row stride E_)
__global__ void h_kernel(const float* __restrict__ z, const float* __restrict__ w1,
                         const float* __restrict__ b1, float* __restrict__ hpre,
                         int dl, int offe) {
    int idx = blockIdx.x * blockDim.x + threadIdx.x;
    if (idx >= M_ * dl) return;
    int m = idx / dl;
    int e = idx - m * dl;
    float acc = b1[e];
    const float* zr = z + m * DZ_;
#pragma unroll
    for (int j = 0; j < DZ_; ++j) acc += zr[j] * w1[j * dl + e];
    acc = (acc >= 0.0f) ? acc : ALPHA_ * acc;
    hpre[(size_t)m * E_ + offe + e] = acc;
}

// per-feature mean / (g * rsqrt(var+eps)) over 2560 rows
__global__ void stats_kernel(const float* __restrict__ hpre, const float* __restrict__ g,
                             float* __restrict__ muA, float* __restrict__ sgA, int offe) {
    __shared__ float rs[256], rs2[256];
    int e = blockIdx.x;
    float s = 0.0f, s2 = 0.0f;
    for (int m = threadIdx.x; m < M_; m += 256) {
        float h = hpre[(size_t)m * E_ + offe + e];
        s += h; s2 += h * h;
    }
    rs[threadIdx.x] = s; rs2[threadIdx.x] = s2;
    __syncthreads();
    for (int st = 128; st > 0; st >>= 1) {
        if (threadIdx.x < st) { rs[threadIdx.x] += rs[threadIdx.x + st]; rs2[threadIdx.x] += rs2[threadIdx.x + st]; }
        __syncthreads();
    }
    if (threadIdx.x == 0) {
        float mu = rs[0] * (1.0f / M_);
        float var = rs2[0] * (1.0f / M_) - mu * mu;
        muA[offe + e] = mu;
        sgA[offe + e] = g[e] * rsqrtf(var + BN_EPS_);
    }
}

// A_bf16 = (h - mu)*sg + x_embed[b]  (row stride E_, col offset offe)
__global__ void norm_kernel(const float* __restrict__ hpre, const float* __restrict__ muA,
                            const float* __restrict__ sgA, const float* __restrict__ xe,
                            __bf16* __restrict__ Abf, int dl, int offe) {
    int idx = blockIdx.x * blockDim.x + threadIdx.x;
    if (idx >= M_ * dl) return;
    int m = idx / dl;
    int e = idx - m * dl;
    int b = m / L_;
    float h = hpre[(size_t)m * E_ + offe + e];
    float v = (h - muA[offe + e]) * sgA[offe + e] + xe[(size_t)b * E_ + offe + e];
    Abf[(size_t)m * E_ + offe + e] = f2bf(v);
}

// w2T[n*dl + k] = bf16(w2[k*dout + n]), tiled transpose
__global__ void w2t_kernel(const float* __restrict__ w2, __bf16* __restrict__ w2T,
                           int dl, int dout) {
    __shared__ __bf16 tile[32][33];
    int kb = blockIdx.y * 32, nb = blockIdx.x * 32;
    for (int j = threadIdx.y; j < 32; j += 8)
        tile[j][threadIdx.x] = f2bf(w2[(size_t)(kb + j) * dout + nb + threadIdx.x]);
    __syncthreads();
    for (int j = threadIdx.y; j < 32; j += 8)
        w2T[(size_t)(nb + j) * dl + kb + threadIdx.x] = tile[threadIdx.x][j];
}

// NB per-(b,d) constants: r = 1/disp, klog = log(maxv*disp),
// cln = lgamma(x+r) - lgamma(r) - lgamma(1+x)
__global__ void nb_kernel(const float* __restrict__ batches, const float* __restrict__ wd,
                          const float* __restrict__ bd, const float* __restrict__ xg,
                          float* __restrict__ rA, float* __restrict__ kA, float* __restrict__ cA,
                          int dout, int xoff, float maxv) {
    int idx = blockIdx.x * blockDim.x + threadIdx.x;
    if (idx >= B_ * dout) return;
    int b = idx / dout;
    int n = idx - b * dout;
    float v = batches[b * 2] * wd[n] + batches[b * 2 + 1] * wd[dout + n] + bd[n];
    float sp = fmaxf(v, 0.0f) + log1pf(__expf(-fabsf(v)));   // softplus
    float disp = fminf(sp, 6.0f);
    float r = 1.0f / disp;
    float x = xg[(size_t)b * D_ + xoff + n];
    rA[idx] = r;
    kA[idx] = logf(maxv * disp);
    cA[idx] = lgammaf(x + r) - lgammaf(r) - lgammaf(1.0f + x);
}

// ---------------- fused WMMA GEMM + masked log-prob reduction ----------------
// grid.x = M_/32; 256 threads = 8 waves. Waves 0-3 -> row-tile 0, 4-7 -> row-tile 1;
// each wave strides over N-tiles by 4, keeps 8 per-lane row partials, atomics at end.
__global__ void gemm_lp_kernel(const __bf16* __restrict__ A, int Aoff,
                               const __bf16* __restrict__ Bt,
                               const float* __restrict__ bias2,
                               const float* __restrict__ xg,
                               const unsigned char* __restrict__ mk,
                               int xoff,
                               const float* __restrict__ rArr,
                               const float* __restrict__ kArr,
                               const float* __restrict__ cArr,
                               int dl, int dout, int isBern,
                               float* __restrict__ out, int outOff) {
    __shared__ __bf16 sA[32 * 528];
    const int LDA = dl + 16;                      // 16B-aligned rows, bank-spread
    const int mBase = blockIdx.x * 32;

    // stage 32 x dl bf16 A tile into LDS (u32 copies)
    for (int t = threadIdx.x; t < (32 * dl) >> 1; t += 256) {
        int idx = t << 1;
        int r = idx / dl;
        int c = idx - r * dl;
        unsigned int v = *(const unsigned int*)(A + (size_t)(mBase + r) * E_ + Aoff + c);
        *(unsigned int*)(&sA[r * LDA + c]) = v;
    }
    __syncthreads();

    const int wave = threadIdx.x >> 5;
    const int lane = threadIdx.x & 31;
    const int l16  = lane & 15;
    const int hi   = lane >> 4;
    const int rowTile = wave >> 2;
    const int mBaseW  = mBase + rowTile * 16;
    const int sRow    = rowTile * 16;

    float racc[8];
#pragma unroll
    for (int v = 0; v < 8; ++v) racc[v] = 0.0f;

    const int ntiles = dout >> 4;
    for (int nt = (wave & 3); nt < ntiles; nt += 4) {
        v8f c = {0.f, 0.f, 0.f, 0.f, 0.f, 0.f, 0.f, 0.f};
        const int n = nt * 16 + l16;
        const __bf16* bcol = Bt + (size_t)n * dl;

        for (int k = 0; k < dl; k += 32) {
            // A fragment: row = l16; K = k + 16*hi + (0..15), contiguous
            const __bf16* ap = &sA[sRow * LDA + l16 * LDA + k + 16 * hi];
            v8bf a0 = *(const v8bf*)(ap);
            v8bf a1 = *(const v8bf*)(ap + 8);
            v16bf af = __builtin_shufflevector(a0, a1, 0,1,2,3,4,5,6,7,8,9,10,11,12,13,14,15);
            // B fragment: col = l16, K = k + 16*hi + (0..15), contiguous
            v16bf bf = *(const v16bf*)(bcol + k + 16 * hi);
            c = __builtin_amdgcn_wmma_f32_16x16x32_bf16(false, af, false, bf,
                                                        (short)0, c, false, false);
        }

        // fused epilogue: C layout — lane holds (M = v + 8*hi, N = l16) for v = 0..7
        const float b2v = bias2[n];
#pragma unroll
        for (int v = 0; v < 8; ++v) {
            const int m = mBaseW + v + 8 * hi;
            const int b = m / L_;
            const size_t gi = (size_t)b * D_ + xoff + n;
            const float x = xg[gi];
            const float o = c[v] + b2v;
            float lp;
            if (isBern) {
                float oc = fminf(fmaxf(o, -11.512925f), 11.512925f); // logit(1e-5)
                float ls = logsig(oc);
                lp = x * ls + (1.0f - x) * (ls - oc);
            } else {
                const int li = b * dout + n;
                float t = logsig(o) + kArr[li];     // logits
                float s = logsig(-t);
                lp = cArr[li] + x * (t + s) + rArr[li] * s;
            }
            racc[v] += mk[gi] ? lp : 0.0f;
        }
    }

    // row-sum across the 16 lanes of each half-wave, then atomics from lanes 0/16
#pragma unroll
    for (int v = 0; v < 8; ++v) {
        float s = racc[v];
        s += __shfl_xor(s, 1);
        s += __shfl_xor(s, 2);
        s += __shfl_xor(s, 4);
        s += __shfl_xor(s, 8);
        racc[v] = s;
    }
    if (l16 == 0) {
#pragma unroll
        for (int v = 0; v < 8; ++v) {
            int m = mBaseW + v + 8 * hi;
            int b = m / L_;
            int l = m - b * L_;
            atomicAdd(&out[b * 15 + outOff + l], racc[v]);
        }
    }
}

// ---------------- host ----------------
extern "C" void kernel_launch(void* const* d_in, const int* in_sizes, int n_in,
                              void* d_out, int out_size, void* d_ws, size_t ws_size,
                              hipStream_t stream) {
    (void)in_sizes; (void)n_in; (void)ws_size;
    const float* x        = (const float*)d_in[0];
    const unsigned char* masks = (const unsigned char*)d_in[2];
    const float* batches  = (const float*)d_in[3];
    const float* z        = (const float*)d_in[4];
    const float* xe       = (const float*)d_in[5];
    const float* w1[3] = {(const float*)d_in[6],  (const float*)d_in[11], (const float*)d_in[16]};
    const float* b1[3] = {(const float*)d_in[7],  (const float*)d_in[12], (const float*)d_in[17]};
    const float* g [3] = {(const float*)d_in[8],  (const float*)d_in[13], (const float*)d_in[18]};
    const float* w2[3] = {(const float*)d_in[9],  (const float*)d_in[14], (const float*)d_in[19]};
    const float* b2[3] = {(const float*)d_in[10], (const float*)d_in[15], (const float*)d_in[20]};
    const float* wd[2] = {(const float*)d_in[21], (const float*)d_in[23]};
    const float* bd[2] = {(const float*)d_in[22], (const float*)d_in[24]};

    const int dl[3]   = {256, 128, 512};
    const int dout[3] = {4000, 256, 16384};
    const int offe[3] = {0, 256, 384};
    const int offx[3] = {0, 4000, 4256};
    const float maxv[2] = {10000.0f, 100.0f};

    // workspace carve-up
    char* base = (char*)d_ws;
    size_t cur = 0;
    auto carve = [&](size_t bytes) { char* p = base + cur; cur += (bytes + 255) & ~(size_t)255; return p; };
    float*  hpre = (float*)carve((size_t)M_ * E_ * 4);
    __bf16* Abf  = (__bf16*)carve((size_t)M_ * E_ * 2);
    float*  muA  = (float*)carve(E_ * 4);
    float*  sgA  = (float*)carve(E_ * 4);
    __bf16* w2T[3];
    for (int i = 0; i < 3; ++i) w2T[i] = (__bf16*)carve((size_t)dl[i] * dout[i] * 2);
    float *rA[2], *kA[2], *cA[2];
    for (int i = 0; i < 2; ++i) {
        rA[i] = (float*)carve((size_t)B_ * dout[i] * 4);
        kA[i] = (float*)carve((size_t)B_ * dout[i] * 4);
        cA[i] = (float*)carve((size_t)B_ * dout[i] * 4);
    }

    zero_kernel<<<(out_size + 255) / 256, 256, 0, stream>>>((float*)d_out, out_size);

    for (int i = 0; i < 3; ++i) {
        int tot = M_ * dl[i];
        h_kernel<<<(tot + 255) / 256, 256, 0, stream>>>(z, w1[i], b1[i], hpre, dl[i], offe[i]);
        stats_kernel<<<dl[i], 256, 0, stream>>>(hpre, g[i], muA, sgA, offe[i]);
        norm_kernel<<<(tot + 255) / 256, 256, 0, stream>>>(hpre, muA, sgA, xe, Abf, dl[i], offe[i]);
        w2t_kernel<<<dim3(dout[i] / 32, dl[i] / 32), dim3(32, 8), 0, stream>>>(w2[i], w2T[i], dl[i], dout[i]);
        if (i < 2) {
            int ntot = B_ * dout[i];
            nb_kernel<<<(ntot + 255) / 256, 256, 0, stream>>>(batches, wd[i], bd[i], x,
                                                              rA[i], kA[i], cA[i],
                                                              dout[i], offx[i], maxv[i]);
        }
    }

    for (int i = 0; i < 3; ++i) {
        int isBern = (i == 2);
        gemm_lp_kernel<<<M_ / 32, 256, 0, stream>>>(
            Abf, offe[i], w2T[i], b2[i], x, masks, offx[i],
            isBern ? nullptr : rA[i], isBern ? nullptr : kA[i], isBern ? nullptr : cA[i],
            dl[i], dout[i], isBern, (float*)d_out, i * 5);
    }
}
